// CYPClassifier_36945308680481
// MI455X (gfx1250) — matrix operand order repaired and checked by reading, not compiled, not run on hardware
//
#include <hip/hip_runtime.h>

typedef __bf16 bf16;
typedef __attribute__((ext_vector_type(16))) __bf16 v16bf;
typedef __attribute__((ext_vector_type(8)))  __bf16 v8bf;
typedef __attribute__((ext_vector_type(8)))  float  v8f;
typedef __attribute__((ext_vector_type(2)))  float  v2f;

#define BATCH 64
#define LSEQ  1024
#define DTGT  1280
#define HID   768
#define NTOT  1536
#define TAPS  9
#define KTOT  (TAPS*DTGT)      /* 11520 */
#define KCHUNKS (KTOT/32)      /* 360 */
#define XROWS 72
#define XPITCH 1288            /* 1280 + 8 bf16 pad -> conflict-free b128 LDS reads */
#define WPITCH 40              /* 32 + 8 bf16 pad */
#define TPITCH 68              /* 64 + 4 f32 pad for transpose staging */
#define WBUFELEMS (192*WPITCH)
/* Xs window + double-buffered weight slabs */
#define SMEM_BYTES (XROWS*XPITCH*2 + 2*WBUFELEMS*2)  /* 216192 B < 320KB WGP LDS */

union BFrag { v16bf v; v8bf h[2]; };

// ---------------------------------------------------------------------------
// CDNA5 async Global->LDS copy (ASYNCcnt path), guarded so the file compiles
// on toolchains lacking the builtins (falls back to a sync 16B copy).
// Builtin signature (from hipcc diagnostic): params are AS-qualified pointers
// to a vector of 4 ints: (v4i __device__*, v4i __shared__*, Ii offset, Ii cpol)
// ---------------------------------------------------------------------------
#if defined(__HIP_DEVICE_COMPILE__) && \
    __has_builtin(__builtin_amdgcn_global_load_async_to_lds_b128) && \
    __has_builtin(__builtin_amdgcn_s_wait_asynccnt)
#define HAVE_ASYNC_LDS 1
#else
#define HAVE_ASYNC_LDS 0
#endif

typedef int v4i_ __attribute__((vector_size(16)));

__device__ __forceinline__ void copy16_g2l(const void* g, void* l)
{
#if HAVE_ASYNC_LDS
  typedef __attribute__((address_space(1))) v4i_ GV;
  typedef __attribute__((address_space(3))) v4i_ LV;
  __builtin_amdgcn_global_load_async_to_lds_b128((GV*)g, (LV*)l, 0, 0);
#else
  *(uint4*)l = *(const uint4*)g;
#endif
}

__device__ __forceinline__ void wait_async_all()
{
#if HAVE_ASYNC_LDS
  __builtin_amdgcn_s_wait_asynccnt(0);
#endif
}

// ---------------------------------------------------------------------------
// K0a: repack feat_w / attn_w (O,C,9) -> bf16 W[n][k], k = tap*1280 + c
// ---------------------------------------------------------------------------
__global__ __launch_bounds__(256) void prep_weights_kernel(
    const float* __restrict__ fw, const float* __restrict__ aw,
    bf16* __restrict__ Wbf)
{
  int idx = blockIdx.x * 256 + threadIdx.x;
  if (idx >= NTOT * KTOT) return;
  int n = idx / KTOT;
  int k = idx - n * KTOT;
  int tap = k / DTGT;
  int c = k - tap * DTGT;
  float v = (n < HID) ? fw[((size_t)n * DTGT + c) * TAPS + tap]
                      : aw[((size_t)(n - HID) * DTGT + c) * TAPS + tap];
  Wbf[idx] = (bf16)v;
}

// ---------------------------------------------------------------------------
// K0c: one-time f32 -> bf16 copy of t_emb (halves recurring HBM traffic and
//      enables async LDS staging in the conv kernel)
// ---------------------------------------------------------------------------
__global__ __launch_bounds__(256) void prep_x_kernel(
    const float* __restrict__ t_emb, bf16* __restrict__ Xbf)
{
  size_t i = ((size_t)blockIdx.x * 256 + threadIdx.x) * 4;
  if (i >= (size_t)BATCH * LSEQ * DTGT) return;
  float4 v = *(const float4*)(t_emb + i);
  bf16* p = Xbf + i;
  p[0] = (bf16)v.x; p[1] = (bf16)v.y; p[2] = (bf16)v.z; p[3] = (bf16)v.w;
}

// ---------------------------------------------------------------------------
// K0b: fold all BN params (+biases) into scale/shift; zero counters/accums
// ---------------------------------------------------------------------------
__global__ __launch_bounds__(256) void prep_params_kernel(
    const float* feat_b, const float* fg, const float* fb, const float* frm, const float* frv,
    const float* attn_b,
    const float* drug_b, const float* dg, const float* db, const float* drm, const float* drv,
    const float* proj_b, const float* pg, const float* pb, const float* prm, const float* prv,
    const float* head1_b, const float* hg, const float* hb, const float* hrm, const float* hrv,
    float* sconv, float* hconv, float* sdrug, float* hdrug,
    float* sproj, float* hproj, float* shead, float* hhead,
    float* cntd, float* cntt, float* accv)
{
  int i = blockIdx.x * 256 + threadIdx.x;
  if (i < HID) {
    float s = fg[i] * rsqrtf(frv[i] + 1e-5f);
    sconv[i] = s;
    hconv[i] = (feat_b[i] - frm[i]) * s + fb[i];
  } else if (i < NTOT) {
    int j = i - HID;
    sconv[i] = 1.f;
    hconv[i] = attn_b[j];
  }
  if (i < 1024) {
    float s = dg[i] * rsqrtf(drv[i] + 1e-5f);
    sdrug[i] = s;
    hdrug[i] = (drug_b[i] - drm[i]) * s + db[i];
    float sp = pg[i] * rsqrtf(prv[i] + 1e-5f);
    sproj[i] = sp;
    hproj[i] = (proj_b[i] - prm[i]) * sp + pb[i];
  }
  if (i < 512) {
    float s = hg[i] * rsqrtf(hrv[i] + 1e-5f);
    shead[i] = s;
    hhead[i] = (head1_b[i] - hrm[i]) * s + hb[i];
    cntd[i] = 0.f;
    cntt[i] = 0.f;
  }
  if (i < 8) accv[i] = 0.f;
}

// ---------------------------------------------------------------------------
// K1: both conv1ds as one implicit GEMM, bf16 WMMA 16x16x32.
//     Block: 8 waves, M=64 (one b, 64 seq positions), N=192 of 1536.
//     Async Global->LDS staging: X window once, weight slab double-buffered
//     (one barrier per K-chunk, fetch overlapped with WMMA).
//     Epilogue: folded BN/bias, LDS transpose, coalesced [b][c][l] stores.
// ---------------------------------------------------------------------------
__global__ __launch_bounds__(256) void conv_wmma_kernel(
    const bf16* __restrict__ Xbf, const bf16* __restrict__ Wbf,
    const float* __restrict__ sc, const float* __restrict__ sh,
    float* __restrict__ convF, float* __restrict__ convA)
{
  extern __shared__ char smem[];
  bf16* Xs  = (bf16*)smem;                    // [72][XPITCH]
  bf16* Ws0 = Xs + XROWS * XPITCH;            // [192][WPITCH] buffer 0
  bf16* Ws1 = Ws0 + WBUFELEMS;                // [192][WPITCH] buffer 1
  float* Ts = (float*)smem;                   // reused after K loop: [192][TPITCH]

  const int tid  = threadIdx.x;
  const int b    = blockIdx.z;
  const int l0   = blockIdx.y * 64;
  const int nblk = blockIdx.x * 192;

  const bf16* wsrc_base = Wbf + (size_t)nblk * KTOT;
  const bf16* xsrc_base = Xbf + (size_t)b * LSEQ * DTGT;

  // Issue async staging of the X window: rows l0-4 .. l0+67, 160 x 16B per row
  for (int i = tid; i < XROWS * 160; i += 256) {
    int r = i / 160, seg = i - r * 160;
    int l = l0 - 4 + r;
    void* ldst = (void*)(Xs + r * XPITCH + seg * 8);
    if (l >= 0 && l < LSEQ)
      copy16_g2l(xsrc_base + (size_t)l * DTGT + seg * 8, ldst);
    else
      *(uint4*)ldst = make_uint4(0u, 0u, 0u, 0u);   // zero halo
  }

  const int lane = tid & 31;
  const int wv   = tid >> 5;
  const int mt   = wv & 3;       // m-tile 0..3 (16 rows each)
  const int nh   = wv >> 2;      // n-half 0..1 (96 cols each)
  const int mrow = lane & 15;
  const int hi   = lane >> 4;
  const int khalfA = hi * 8;     // A: lanes>=16 hold K 8..15 / 24..31
  const int khalfB = hi * 8;     // B: lanes>=16 hold K 8..15 / 24..31

  v8f acc[6];
#pragma unroll
  for (int t = 0; t < 6; t++)
#pragma unroll
    for (int e = 0; e < 8; e++) acc[t][e] = 0.f;

  // Prologue: stage weight slab for chunk 0 into buffer 0
  {
    const bf16* wsrc = wsrc_base;
    for (int i = tid; i < 192 * 4; i += 256) {
      int nn = i >> 2, k8 = (i & 3) * 8;
      copy16_g2l(wsrc + (size_t)nn * KTOT + k8, Ws0 + nn * WPITCH + k8);
    }
  }

  for (int kc = 0; kc < KCHUNKS; ++kc) {
    bf16* cur = (kc & 1) ? Ws1 : Ws0;
    bf16* nxt = (kc & 1) ? Ws0 : Ws1;

    wait_async_all();     // my async writes for 'cur' (and X on kc==0) landed
    __syncthreads();      // everyone's writes landed; everyone done with 'nxt'

    if (kc + 1 < KCHUNKS) {   // prefetch next slab while computing this one
      const bf16* wsrc = wsrc_base + (kc + 1) * 32;
      for (int i = tid; i < 192 * 4; i += 256) {
        int nn = i >> 2, k8 = (i & 3) * 8;
        copy16_g2l(wsrc + (size_t)nn * KTOT + k8, nxt + nn * WPITCH + k8);
      }
    }

    const int tap   = kc / 40;
    const int cbase = (kc % 40) * 32;

    // A fragment: row l0 + mt*16 + m + tap - 4  ->  Xs row mt*16 + m + tap
    BFrag Af;
    const bf16* ap = Xs + (mt * 16 + mrow + tap) * XPITCH + cbase + khalfA;
    Af.h[0] = *(const v8bf*)(ap);
    Af.h[1] = *(const v8bf*)(ap + 16);
#pragma unroll
    for (int t = 0; t < 6; t++) {
      BFrag Bf;
      const bf16* bp = cur + (nh * 96 + t * 16 + mrow) * WPITCH + khalfB;
      Bf.h[0] = *(const v8bf*)(bp);
      Bf.h[1] = *(const v8bf*)(bp + 16);
      acc[t] = __builtin_amdgcn_wmma_f32_16x16x32_bf16(
          false, Af.v, false, Bf.v, (short)0, acc[t], false, false);
    }
  }
  wait_async_all();
  __syncthreads();

  // Fold BN/bias, transpose through LDS: Ts[n_local][m]
#pragma unroll
  for (int t = 0; t < 6; t++) {
    int nl = nh * 96 + t * 16 + mrow;
    int n  = nblk + nl;
    float s = sc[n], b0 = sh[n];
    int mb = mt * 16 + hi * 8;
#pragma unroll
    for (int r = 0; r < 8; r++)
      Ts[nl * TPITCH + mb + r] = acc[t][r] * s + b0;
  }
  __syncthreads();

  // Coalesced stores: conv[b][c][l0..l0+63]
  for (int i = tid; i < 192 * 16; i += 256) {
    int nl = i >> 4, o4 = (i & 15) * 4;
    int n = nblk + nl;
    float4 v = *(const float4*)(Ts + nl * TPITCH + o4);
    float* dst = (n < HID)
        ? (convF + ((size_t)(b * HID + n)) * LSEQ + l0 + o4)
        : (convA + ((size_t)(b * HID + (n - HID))) * LSEQ + l0 + o4);
    *(float4*)dst = v;
  }
}

// ---------------------------------------------------------------------------
// K2: per (b, channel): masked softmax-weighted sum + masked max over L
// ---------------------------------------------------------------------------
__global__ __launch_bounds__(256) void pool_kernel(
    const float* __restrict__ convF, const float* __restrict__ convA,
    const unsigned char* __restrict__ mask, float* __restrict__ comb)
{
  __shared__ float r1[256], r2[256];
  const int b = blockIdx.y, c = blockIdx.x, tid = threadIdx.x;
  const float* f = convF + ((size_t)(b * HID + c)) * LSEQ;
  const float* a = convA + ((size_t)(b * HID + c)) * LSEQ;
  const unsigned char* mk = mask + (size_t)b * LSEQ;

  float av[4], fv[4];
  float amax = -3.4028235e38f, gmax = -3.4028235e38f;
#pragma unroll
  for (int j = 0; j < 4; j++) {
    int l = tid + j * 256;
    bool m = mk[l] != 0;
    float aa = m ? a[l] : -1.0e9f;
    float ff = f[l];
    av[j] = aa; fv[j] = ff;
    amax = fmaxf(amax, aa);
    if (m) gmax = fmaxf(gmax, ff);
  }
  r1[tid] = amax; r2[tid] = gmax; __syncthreads();
  for (int s = 128; s > 0; s >>= 1) {
    if (tid < s) { r1[tid] = fmaxf(r1[tid], r1[tid + s]); r2[tid] = fmaxf(r2[tid], r2[tid + s]); }
    __syncthreads();
  }
  const float amaxT = r1[0];
  const float gmaxT = r2[0];
  __syncthreads();

  float se = 0.f, wsum = 0.f;
#pragma unroll
  for (int j = 0; j < 4; j++) { float e = expf(av[j] - amaxT); se += e; wsum += fv[j] * e; }
  r1[tid] = se; r2[tid] = wsum; __syncthreads();
  for (int s = 128; s > 0; s >>= 1) {
    if (tid < s) { r1[tid] += r1[tid + s]; r2[tid] += r2[tid + s]; }
    __syncthreads();
  }
  if (tid == 0) {
    comb[b * NTOT + c]       = r2[0] / r1[0];  // weighted
    comb[b * NTOT + HID + c] = gmaxT;          // gmax
  }
}

// ---------------------------------------------------------------------------
// K3: fp32 WMMA GEMM (16x16x4) + folded BN + leaky. out[m][n] = A[m]·W[n]
// ---------------------------------------------------------------------------
__global__ __launch_bounds__(256) void gemm_f32_wmma(
    const float* __restrict__ A, const float* __restrict__ W,
    const float* __restrict__ sc, const float* __restrict__ sh,
    float* __restrict__ out, int K, int N)
{
  const int lane = threadIdx.x & 31;
  const int wv   = threadIdx.x >> 5;
  const int n0 = blockIdx.x * 128 + wv * 16;
  const int m0 = blockIdx.y * 16;
  const int mr = lane & 15;
  const int klo = (lane >> 4) * 2;   // lanes>=16 hold K+2, K+3

  v8f acc;
#pragma unroll
  for (int e = 0; e < 8; e++) acc[e] = 0.f;

  const float* ap = A + (size_t)(m0 + mr) * K + klo;
  const float* bp = W + (size_t)(n0 + mr) * K + klo;
  for (int k = 0; k < K; k += 4) {
    v2f a; a[0] = ap[k]; a[1] = ap[k + 1];
    v2f bb; bb[0] = bp[k]; bb[1] = bp[k + 1];
    acc = __builtin_amdgcn_wmma_f32_16x16x4_f32(
        false, a, false, bb, (short)0, acc, false, false);
  }
  const int n = n0 + mr;
  const float s = sc[n], b0 = sh[n];
#pragma unroll
  for (int r = 0; r < 8; r++) {
    int m = m0 + r + (lane >> 4) * 8;
    float v = acc[r] * s + b0;
    v = (v >= 0.f) ? v : 0.01f * v;   // leaky
    out[(size_t)m * N + n] = v;
  }
}

// ---------------------------------------------------------------------------
// K4: vector quantization: argmin over 512 codes, counts, loss, gather q->h
// ---------------------------------------------------------------------------
__global__ __launch_bounds__(256) void vq_kernel(
    const float* __restrict__ Z, const float* __restrict__ CB,
    float* __restrict__ h, int col_off,
    float* __restrict__ cnt, float* __restrict__ loss_acc)
{
  __shared__ float zs[64];
  __shared__ float bd[256];
  __shared__ int   bi[256];
  const int tid = threadIdx.x, r = blockIdx.x;
  if (tid < 64) zs[tid] = Z[r * 64 + tid];
  __syncthreads();

  float best = 3.4028235e38f; int bidx = 0;
  for (int j = tid; j < 512; j += 256) {
    const float* cb = CB + j * 64;
    float d = 0.f;
#pragma unroll 8
    for (int k = 0; k < 64; k++) { float t = zs[k] - cb[k]; d += t * t; }
    if (d < best) { best = d; bidx = j; }
  }
  bd[tid] = best; bi[tid] = bidx; __syncthreads();
  for (int s = 128; s > 0; s >>= 1) {
    if (tid < s) {
      if (bd[tid + s] < bd[tid] ||
          (bd[tid + s] == bd[tid] && bi[tid + s] < bi[tid])) {
        bd[tid] = bd[tid + s]; bi[tid] = bi[tid + s];
      }
    }
    __syncthreads();
  }
  if (tid == 0) {
    atomicAdd(cnt + bi[0], 1.f);
    atomicAdd(loss_acc, bd[0] * (1.f / 65536.f));  // mean over 1024*64 elems
  }
  const int idx = bi[0];
  if (tid < 64) {
    int bb = r >> 4, rr = r & 15;
    h[bb * 2048 + col_off + rr * 64 + tid] = CB[idx * 64 + tid];
  }
}

// ---------------------------------------------------------------------------
// K5: head2 logits + evidential loss (digamma/gammaln on device)
// ---------------------------------------------------------------------------
__device__ inline float digamma_f(float x)
{
  float r = 0.f;
  while (x < 6.f) { r -= 1.f / x; x += 1.f; }
  float iv = 1.f / x, iv2 = iv * iv;
  return r + logf(x) - 0.5f * iv -
         iv2 * (0.0833333333f - iv2 * (0.0083333333f - iv2 * 0.0039682540f));
}

__global__ void head2_edl_kernel(
    const float* __restrict__ h1, const float* __restrict__ w2,
    const float* __restrict__ b2, const int* __restrict__ y,
    float* __restrict__ accv)
{
  __shared__ float red[64];
  const int i = threadIdx.x;  // 64 threads, one batch row each
  float al[2];
#pragma unroll
  for (int c = 0; c < 2; c++) {
    float s = b2[c];
    const float* hr = h1 + i * 512;
    const float* wr = w2 + c * 512;
    for (int k = 0; k < 512; k++) s += hr[k] * wr[k];
    float sp = (s > 0.f) ? (s + log1pf(expf(-s))) : log1pf(expf(s));
    al[c] = sp + 1.f;  // alpha = softplus(logit) + 1
  }
  const int yi = y[i];
  float S  = al[0] + al[1];
  float ce = digamma_f(S) - digamma_f(yi == 0 ? al[0] : al[1]);
  float at0 = (yi == 0) ? 1.f : al[0];
  float at1 = (yi == 1) ? 1.f : al[1];
  float St  = at0 + at1;
  float dSt = digamma_f(St);
  float kl = lgammaf(St) - lgammaf(at0) - lgammaf(at1) - lgammaf(2.f) +
             (at0 - 1.f) * (digamma_f(at0) - dSt) +
             (at1 - 1.f) * (digamma_f(at1) - dSt);
  red[i] = ce + 0.1f * kl;   // EVD_LAMBDA = 0.1
  __syncthreads();
  for (int s = 32; s > 0; s >>= 1) { if (i < s) red[i] += red[i + s]; __syncthreads(); }
  if (i == 0) accv[2] = red[0] * (1.f / 64.f);
}

// ---------------------------------------------------------------------------
// K6: perplexities + final 6 scalars
// ---------------------------------------------------------------------------
__global__ __launch_bounds__(256) void finalize_kernel(
    const float* __restrict__ cntd, const float* __restrict__ cntt,
    const float* __restrict__ accv, float* __restrict__ out)
{
  __shared__ float r1[256], r2[256];
  const int tid = threadIdx.x;
  float sd = 0.f, st = 0.f;
  for (int j = tid; j < 512; j += 256) {
    float ad = cntd[j] * (1.f / 1024.f); sd += ad * logf(ad + 1e-10f);
    float at = cntt[j] * (1.f / 1024.f); st += at * logf(at + 1e-10f);
  }
  r1[tid] = sd; r2[tid] = st; __syncthreads();
  for (int s = 128; s > 0; s >>= 1) {
    if (tid < s) { r1[tid] += r1[tid + s]; r2[tid] += r2[tid + s]; }
    __syncthreads();
  }
  if (tid == 0) {
    float dperp = expf(-r1[0]), tperp = expf(-r2[0]);
    float cl = accv[2], dv = accv[0], tv = accv[1];
    out[0] = cl + dv + tv;
    out[1] = cl;
    out[2] = dv;
    out[3] = tv;
    out[4] = dperp;
    out[5] = tperp;
  }
}

// ---------------------------------------------------------------------------
extern "C" void kernel_launch(void* const* d_in, const int* in_sizes, int n_in,
                              void* d_out, int out_size, void* d_ws, size_t ws_size,
                              hipStream_t stream)
{
  const float* d_emb   = (const float*)d_in[0];
  const float* t_emb   = (const float*)d_in[1];
  const unsigned char* t_mask = (const unsigned char*)d_in[2];
  const int*   y       = (const int*)d_in[3];
  const float* drug_w  = (const float*)d_in[4];
  const float* drug_b  = (const float*)d_in[5];
  const float* drug_g  = (const float*)d_in[6];
  const float* drug_bb = (const float*)d_in[7];
  const float* drug_rm = (const float*)d_in[8];
  const float* drug_rv = (const float*)d_in[9];
  const float* feat_w  = (const float*)d_in[10];
  const float* feat_b  = (const float*)d_in[11];
  const float* feat_g  = (const float*)d_in[12];
  const float* feat_bb = (const float*)d_in[13];
  const float* feat_rm = (const float*)d_in[14];
  const float* feat_rv = (const float*)d_in[15];
  const float* attn_w  = (const float*)d_in[16];
  const float* attn_b  = (const float*)d_in[17];
  const float* proj_w  = (const float*)d_in[18];
  const float* proj_b  = (const float*)d_in[19];
  const float* proj_g  = (const float*)d_in[20];
  const float* proj_bb = (const float*)d_in[21];
  const float* proj_rm = (const float*)d_in[22];
  const float* proj_rv = (const float*)d_in[23];
  const float* cb_d    = (const float*)d_in[24];
  const float* cb_t    = (const float*)d_in[25];
  const float* head1_w = (const float*)d_in[26];
  const float* head1_b = (const float*)d_in[27];
  const float* head_g  = (const float*)d_in[28];
  const float* head_bb = (const float*)d_in[29];
  const float* head_rm = (const float*)d_in[30];
  const float* head_rv = (const float*)d_in[31];
  const float* head2_w = (const float*)d_in[32];
  const float* head2_b = (const float*)d_in[33];

  char* ws = (char*)d_ws;
  size_t off = 0;
  bf16*  WBF   = (bf16*)(ws + off);  off += (size_t)NTOT * KTOT * 2;
  bf16*  XBF   = (bf16*)(ws + off);  off += (size_t)BATCH * LSEQ * DTGT * 2;
  float* CONVF = (float*)(ws + off); off += (size_t)BATCH * HID * LSEQ * 4;
  float* CONVA = (float*)(ws + off); off += (size_t)BATCH * HID * LSEQ * 4;
  float* COMB  = (float*)(ws + off); off += (size_t)BATCH * NTOT * 4;
  float* DZ    = (float*)(ws + off); off += (size_t)BATCH * 1024 * 4;
  float* TZ    = (float*)(ws + off); off += (size_t)BATCH * 1024 * 4;
  float* H     = (float*)(ws + off); off += (size_t)BATCH * 2048 * 4;
  float* H1    = (float*)(ws + off); off += (size_t)BATCH * 512 * 4;
  float* SCONV = (float*)(ws + off); off += NTOT * 4;
  float* HCONV = (float*)(ws + off); off += NTOT * 4;
  float* SDRUG = (float*)(ws + off); off += 1024 * 4;
  float* HDRUG = (float*)(ws + off); off += 1024 * 4;
  float* SPROJ = (float*)(ws + off); off += 1024 * 4;
  float* HPROJ = (float*)(ws + off); off += 1024 * 4;
  float* SHEAD = (float*)(ws + off); off += 512 * 4;
  float* HHEAD = (float*)(ws + off); off += 512 * 4;
  float* CNTD  = (float*)(ws + off); off += 512 * 4;
  float* CNTT  = (float*)(ws + off); off += 512 * 4;
  float* ACC   = (float*)(ws + off); off += 32;

  prep_weights_kernel<<<(NTOT * KTOT + 255) / 256, 256, 0, stream>>>(feat_w, attn_w, WBF);
  prep_x_kernel<<<(BATCH * LSEQ * DTGT / 4 + 255) / 256, 256, 0, stream>>>(t_emb, XBF);
  prep_params_kernel<<<6, 256, 0, stream>>>(
      feat_b, feat_g, feat_bb, feat_rm, feat_rv, attn_b,
      drug_b, drug_g, drug_bb, drug_rm, drug_rv,
      proj_b, proj_g, proj_bb, proj_rm, proj_rv,
      head1_b, head_g, head_bb, head_rm, head_rv,
      SCONV, HCONV, SDRUG, HDRUG, SPROJ, HPROJ, SHEAD, HHEAD, CNTD, CNTT, ACC);

  conv_wmma_kernel<<<dim3(8, 16, 64), 256, SMEM_BYTES, stream>>>(
      XBF, WBF, SCONV, HCONV, CONVF, CONVA);
  pool_kernel<<<dim3(HID, BATCH), 256, 0, stream>>>(CONVF, CONVA, t_mask, COMB);

  gemm_f32_wmma<<<dim3(8, 4), 256, 0, stream>>>(d_emb, drug_w, SDRUG, HDRUG, DZ, 768, 1024);
  gemm_f32_wmma<<<dim3(8, 4), 256, 0, stream>>>(COMB,  proj_w, SPROJ, HPROJ, TZ, 1536, 1024);

  vq_kernel<<<1024, 256, 0, stream>>>(DZ, cb_d, H, 0,    CNTD, ACC + 0);
  vq_kernel<<<1024, 256, 0, stream>>>(TZ, cb_t, H, 1024, CNTT, ACC + 1);

  gemm_f32_wmma<<<dim3(4, 4), 256, 0, stream>>>(H, head1_w, SHEAD, HHEAD, H1, 2048, 512);

  head2_edl_kernel<<<1, 64, 0, stream>>>(H1, head2_w, head2_b, y, ACC);
  finalize_kernel<<<1, 256, 0, stream>>>(CNTD, CNTT, ACC, (float*)d_out);
}
